// NetworkActivity_layer_52561809768579
// MI455X (gfx1250) — compile-verified, matrix operand order checked by reading, not run on hardware
//
#include <hip/hip_runtime.h>

// ---------------------------------------------------------------------------
// out(B=4096, P=500) = x(B,G=15000) @ (weight(P,G) * mask(G,P)^T)^T + bias(P)
//
// Memory-bound GEMM (AI ~195 FLOP/B; HBM floor ~13.5us @ 23.3 TB/s; x = 246MB
// dominates, weight+mask = 60MB stay resident in the 192MB L2). FP32 inputs
// are converted to F16 once per tile and fed to v_wmma_f32_16x16x32_f16 with
// f32 accumulation. Double-buffered LDS: global loads of tile t+1 are issued
// before compute of tile t so HBM latency overlaps the WMMA stream. The K
// tail (15000 = 234*64 + 24) is peeled via a uniform scalar branch so the
// steady-state loop carries no exec-mask guards.
// ---------------------------------------------------------------------------

typedef __attribute__((ext_vector_type(16))) _Float16 v16h;
typedef __attribute__((ext_vector_type(8)))  _Float16 v8h;
typedef __attribute__((ext_vector_type(8)))  float    v8f;

#define TILE_M 128
#define TILE_N 64
#define TILE_K 64
#define PAD_K  8                 // 16B pad per LDS row (bank spread)
#define LDK    (TILE_K + PAD_K)  // 72 halfs = 144B, 16B aligned

#define DIM_B 4096
#define DIM_G 15000
#define DIM_P 500
#define NTILE ((DIM_G + TILE_K - 1) / TILE_K)   // 235

// v16h fragment from an LDS row per CDNA5 16-bit 16x32 layout: lane-half h
// holds K = h*8+0..7 (elems 0..7) and K = 16+h*8+0..7 (elems 8..15).
__device__ __forceinline__ v16h ldfrag(const _Float16* p) {
    v8h lo = *(const v8h*)(p);
    v8h hi = *(const v8h*)(p + 16);
    v16h f;
#pragma unroll
    for (int i = 0; i < 8; ++i) { f[i] = lo[i]; f[i + 8] = hi[i]; }
    return f;
}

// Issue global loads for one K-tile into registers. GUARD only for the tail.
template <bool GUARD>
__device__ __forceinline__ void stage_load(const float* __restrict__ x,
                                           const float* __restrict__ weight,
                                           const float* __restrict__ mask,
                                           int tid, int m0, int n0, int k0,
                                           float4 xr[8], float4 wr[4], float mk[4][4]) {
#pragma unroll
    for (int i = 0; i < 8; ++i) {                 // x tile: 128x64, 8 float4/thr
        const int flat = tid + i * 256;
        const int row  = flat >> 4;
        const int col  = (flat & 15) * 4;
        if (!GUARD || (k0 + col < DIM_G))
            xr[i] = *(const float4*)(x + (m0 + row) * DIM_G + k0 + col);
        else
            xr[i] = make_float4(0.f, 0.f, 0.f, 0.f);
    }
#pragma unroll
    for (int i = 0; i < 4; ++i) {                 // w'/mask tile: 64x64
        const int flat = tid + i * 256;
        const int row  = flat >> 4;
        const int col  = (flat & 15) * 4;
        const int p    = min(n0 + row, DIM_P - 1);  // branch-free N clamp
        const int g    = k0 + col;
        if (!GUARD || (g < DIM_G)) {
            wr[i] = *(const float4*)(weight + p * DIM_G + g);
#pragma unroll
            for (int j = 0; j < 4; ++j) mk[i][j] = mask[(g + j) * DIM_P + p];
        } else {
            wr[i] = make_float4(0.f, 0.f, 0.f, 0.f);
#pragma unroll
            for (int j = 0; j < 4; ++j) mk[i][j] = 0.f;
        }
    }
}

// Convert staged registers to f16 and store into an LDS buffer pair.
__device__ __forceinline__ void stage_store(_Float16* __restrict__ sA,
                                            _Float16* __restrict__ sB, int tid,
                                            const float4 xr[8], const float4 wr[4],
                                            const float mk[4][4]) {
#pragma unroll
    for (int i = 0; i < 8; ++i) {
        const int flat = tid + i * 256;
        _Float16* d = &sA[(flat >> 4) * LDK + (flat & 15) * 4];
        d[0] = (_Float16)xr[i].x; d[1] = (_Float16)xr[i].y;
        d[2] = (_Float16)xr[i].z; d[3] = (_Float16)xr[i].w;
    }
#pragma unroll
    for (int i = 0; i < 4; ++i) {
        const int flat = tid + i * 256;
        _Float16* d = &sB[(flat >> 4) * LDK + (flat & 15) * 4];
        d[0] = (_Float16)(wr[i].x * mk[i][0]);
        d[1] = (_Float16)(wr[i].y * mk[i][1]);
        d[2] = (_Float16)(wr[i].z * mk[i][2]);
        d[3] = (_Float16)(wr[i].w * mk[i][3]);
    }
}

// One K-tile of compute: two K=32 sub-steps, 8 WMMAs per wave.
__device__ __forceinline__ void compute_tile(const _Float16* __restrict__ sA,
                                             const _Float16* __restrict__ sB,
                                             int waveM, int waveN, int l16, int kh,
                                             v8f acc[2][2]) {
#pragma unroll
    for (int ks = 0; ks < TILE_K; ks += 32) {
        v16h a[2], b[2];
#pragma unroll
        for (int mt = 0; mt < 2; ++mt)
            a[mt] = ldfrag(&sA[(waveM * 32 + mt * 16 + l16) * LDK + ks + kh * 8]);
#pragma unroll
        for (int nt = 0; nt < 2; ++nt)
            b[nt] = ldfrag(&sB[(waveN * 32 + nt * 16 + l16) * LDK + ks + kh * 8]);
#pragma unroll
        for (int mt = 0; mt < 2; ++mt)
#pragma unroll
            for (int nt = 0; nt < 2; ++nt)
                acc[mt][nt] = __builtin_amdgcn_wmma_f32_16x16x32_f16(
                    /*neg_a=*/false, a[mt], /*neg_b=*/false, b[nt],
                    /*c_mod=*/(short)0, acc[mt][nt],
                    /*reuse_a=*/false, /*reuse_b=*/false);
    }
}

__global__ __launch_bounds__(256)
void pathway_masked_gemm_wmma(const float* __restrict__ x,
                              const float* __restrict__ weight,
                              const float* __restrict__ mask,
                              const float* __restrict__ bias,
                              float* __restrict__ out) {
    __shared__ _Float16 sA[2][TILE_M * LDK];   // 2 x 18KB
    __shared__ _Float16 sB[2][TILE_N * LDK];   // 2 x  9KB

    const int tid   = threadIdx.x;
    const int wave  = tid >> 5;
    const int lane  = tid & 31;
    const int kh    = lane >> 4;
    const int l16   = lane & 15;
    const int waveM = wave & 3;        // 4 waves along M -> 128
    const int waveN = wave >> 2;       // 2 waves along N -> 64
    const int m0    = blockIdx.x * TILE_M;
    const int n0    = blockIdx.y * TILE_N;

    v8f acc[2][2] = {};
    float4 xr[8];
    float4 wr[4];
    float  mk[4][4];

    // prologue: tile 0 (always full) staged into buffer 0
    stage_load<false>(x, weight, mask, tid, m0, n0, 0, xr, wr, mk);
    stage_store(sA[0], sB[0], tid, xr, wr, mk);
    __syncthreads();

    int buf = 0;
    for (int t = 0; t < NTILE; ++t) {
        const int kn = (t + 1) * TILE_K;
        const bool has_next = kn < DIM_G;
        if (has_next) {
            if (kn + TILE_K <= DIM_G)             // uniform scalar branch
                stage_load<false>(x, weight, mask, tid, m0, n0, kn, xr, wr, mk);
            else                                  // peeled tail (last tile only)
                stage_load<true>(x, weight, mask, tid, m0, n0, kn, xr, wr, mk);
            // prefetch tile t+2 of x (global_prefetch_b8), covers 128B segments
            if (kn + TILE_K < DIM_G) {
                const int row = tid >> 1;
                const int seg = (tid & 1) * 32;
                __builtin_prefetch(x + (m0 + row) * DIM_G + kn + TILE_K + seg, 0, 1);
            }
        }
        compute_tile(sA[buf], sB[buf], waveM, waveN, l16, kh, acc);
        if (has_next) {
            stage_store(sA[buf ^ 1], sB[buf ^ 1], tid, xr, wr, mk);
            __syncthreads();
            buf ^= 1;
        }
    }

    // epilogue: C layout = VGPR r, lane -> M = r + 8*(lane>>4), N = lane&15
#pragma unroll
    for (int nt = 0; nt < 2; ++nt) {
        const int n = n0 + waveN * 32 + nt * 16 + l16;
        if (n >= DIM_P) continue;
        const float bv = bias[n];
#pragma unroll
        for (int mt = 0; mt < 2; ++mt) {
            const int mbase = m0 + waveM * 32 + mt * 16 + kh * 8;
#pragma unroll
            for (int r = 0; r < 8; ++r)
                out[(mbase + r) * DIM_P + n] = acc[mt][nt][r] + bv;
        }
    }
}

extern "C" void kernel_launch(void* const* d_in, const int* in_sizes, int n_in,
                              void* d_out, int out_size, void* d_ws, size_t ws_size,
                              hipStream_t stream) {
    const float* x      = (const float*)d_in[0];  // (4096, 15000)
    const float* weight = (const float*)d_in[1];  // (500, 15000)
    const float* mask   = (const float*)d_in[2];  // (15000, 500)
    const float* bias   = (const float*)d_in[3];  // (500,)
    float* out          = (float*)d_out;          // (4096, 500)

    dim3 grid(DIM_B / TILE_M, (DIM_P + TILE_N - 1) / TILE_N);  // 32 x 8 = 256 blocks
    dim3 block(256);
    pathway_masked_gemm_wmma<<<grid, block, 0, stream>>>(x, weight, mask, bias, out);
}
